// DotProductAttention_58686433132651
// MI455X (gfx1250) — matrix operand order repaired and checked by reading, not compiled, not run on hardware
//
#include <hip/hip_runtime.h>

typedef __attribute__((ext_vector_type(16))) _Float16 v16h;
typedef __attribute__((ext_vector_type(8)))  _Float16 v8h;
typedef __attribute__((ext_vector_type(2)))  __fp16   h2;   // cvt_pkrtz result type
typedef __attribute__((ext_vector_type(8)))  float    v8f;
typedef __attribute__((ext_vector_type(4)))  float    f32x4;

#define B_   16
#define LQ_  2048
#define LK_  2048
#define D_   64
#define TN   64      // KV tile
#define PSTR 72      // padded LDS row stride (halves): 144B, 16B-aligned
// Q is pre-scaled by (1/sqrt(D)) * log2(e) so the main loop uses raw v_exp_f32.
#define QSCALE 0.1803368801111f

union U16 { v16h v; v8h h[2]; };
union U8  { v8h  v; h2  p[4]; };

// ---------------- prologue: f32 -> f16 convert (optionally scaled) ----------
__global__ __launch_bounds__(256)
void cvt_f32_f16(const float* __restrict__ src, _Float16* __restrict__ dst, float scale) {
    const int i = (blockIdx.x * 256 + threadIdx.x) * 8;
    f32x4 a = *(const f32x4*)(src + i);
    f32x4 b = *(const f32x4*)(src + i + 4);
    U8 u;
    u.p[0] = __builtin_amdgcn_cvt_pkrtz(a[0] * scale, a[1] * scale);
    u.p[1] = __builtin_amdgcn_cvt_pkrtz(a[2] * scale, a[3] * scale);
    u.p[2] = __builtin_amdgcn_cvt_pkrtz(b[0] * scale, b[1] * scale);
    u.p[3] = __builtin_amdgcn_cvt_pkrtz(b[2] * scale, b[3] * scale);
    *(v8h*)(dst + i) = u.v;
}

// ---------------- prologue: V [b][n][d] f32 -> Vt [b][d][n] f16 -------------
__global__ __launch_bounds__(256)
void transpose_v(const float* __restrict__ V, _Float16* __restrict__ Vt) {
    __shared__ _Float16 T[D_ * PSTR];
    const int tid = threadIdx.x;
    const int b   = blockIdx.x >> 5;          // / (LK/64)
    const int n0  = (blockIdx.x & 31) * 64;
    const float* vb = V + ((size_t)b * LK_ + n0) * D_;
#pragma unroll
    for (int it = 0; it < 4; ++it) {          // 1024 float4s, coalesced reads
        const int idx = tid + it * 256;
        const int n = idx >> 4;
        const int c = (idx & 15) * 4;
        f32x4 v = *(const f32x4*)(vb + n * D_ + c);
#pragma unroll
        for (int j = 0; j < 4; ++j) T[(c + j) * PSTR + n] = (_Float16)v[j];
    }
    __syncthreads();
    _Float16* ob = Vt + (size_t)b * D_ * LK_ + n0;
#pragma unroll
    for (int it = 0; it < 2; ++it) {          // coalesced 16B writes
        const int idx = tid + it * 256;
        const int d = idx >> 3;
        const int c = (idx & 7) * 8;
        *(v8h*)(ob + (size_t)d * LK_ + c) = *(const v8h*)&T[d * PSTR + c];
    }
}

// ---------------- main: flash attention, transposed tiles -------------------
// Per wave: 16 query rows (column dim of S^T).  S^T = K_tile x Q^T (A=K, B=Q^T),
// O^T = V^T x P^T (A=V^T rows, B=P^T).  Lane holds query column m = lane&15.
__global__ __launch_bounds__(256)
void fa_fwd(const _Float16* __restrict__ Qh, const _Float16* __restrict__ Kh,
            const _Float16* __restrict__ Vt, const int* __restrict__ VL,
            float* __restrict__ O)
{
    __shared__ _Float16 Psh[8 * 16 * PSTR];   // per-wave P staging [m][n]

    const int tid  = threadIdx.x;
    const int wave = tid >> 5;
    const int lane = tid & 31;
    const int lh   = lane & 15;
    const int hi   = lane >> 4;

    const int batch = blockIdx.x >> 4;             // / (LQ/128)
    const int qbase = (blockIdx.x & 15) * 128;     // 8 waves x 16 rows
    const int valid = VL[batch];
    const int kmax  = min(LK_, (valid + TN - 1) & ~(TN - 1));  // uniform

    // Q^T B-fragments, persistent: col m = qbase + wave*16 + lh
    const _Float16* qrow = Qh + ((size_t)batch * LQ_ + qbase + wave * 16 + lh) * D_;
    U16 qb[2];
#pragma unroll
    for (int kt = 0; kt < 2; ++kt) {
        const _Float16* p = qrow + kt * 32 + hi * 16;   // contiguous 32B
        qb[kt].h[0] = *(const v8h*)(p);
        qb[kt].h[1] = *(const v8h*)(p + 8);
    }

    const _Float16* kbase = Kh + (size_t)batch * LK_ * D_;
    const _Float16* vbase = Vt + (size_t)batch * D_ * LK_;
    _Float16* Pw = &Psh[wave * 16 * PSTR];

    v8f oacc[4];                                   // O^T: elem i <-> d = dt*16+hi*8+i
#pragma unroll
    for (int dt = 0; dt < 4; ++dt) oacc[dt] = (v8f){};
    float mrun = -3.0e38f, lrun = 0.0f;

    for (int kv = 0; kv < kmax; kv += TN) {
        if (kv + TN < kmax) {                      // prefetch next KV tile (WGP scope)
            __builtin_prefetch(kbase + (size_t)(kv + TN + lane) * D_, 0, 3);
            __builtin_prefetch(kbase + (size_t)(kv + TN + 32 + lane) * D_, 0, 3);
            __builtin_prefetch(vbase + (size_t)lane * LK_ + kv + TN, 0, 3);
            __builtin_prefetch(vbase + (size_t)(lane + 32) * LK_ + kv + TN, 0, 3);
        }

        // ---- S^T tiles: A = K rows (n), B = Q^T -----------------------------
        v8f s[4];
#pragma unroll
        for (int nt = 0; nt < 4; ++nt) {
            const _Float16* kr = kbase + (size_t)(kv + nt * 16 + lh) * D_;
            v8f acc = (v8f){};
#pragma unroll
            for (int kt = 0; kt < 2; ++kt) {
                U16 ka;                            // A-frag: two 16B chunks
                const _Float16* c0 = kr + kt * 32 + hi * 16;
                ka.h[0] = *(const v8h*)(c0);
                ka.h[1] = *(const v8h*)(c0 + 8);
                acc = __builtin_amdgcn_wmma_f32_16x16x32_f16(
                        false, ka.v, false, qb[kt].v, (short)0, acc, false, false);
            }
            s[nt] = acc;                           // elem i <-> n = nt*16+hi*8+i, col m
        }

        // ---- mask only on the boundary tile (reference: -1e6) ---------------
        if (kv + TN > valid) {
#pragma unroll
            for (int nt = 0; nt < 4; ++nt) {
                const int nb = kv + nt * 16 + hi * 8;
#pragma unroll
                for (int i = 0; i < 8; ++i)
                    if (nb + i >= valid) s[nt][i] = -1.0e6f;
            }
        }

        // ---- row (=column m) max: tree + ONE cross-half exchange ------------
        v8f t;
#pragma unroll
        for (int i = 0; i < 8; ++i)
            t[i] = fmaxf(fmaxf(s[0][i], s[1][i]), fmaxf(s[2][i], s[3][i]));
        float tm = t[0];
#pragma unroll
        for (int i = 1; i < 8; ++i) tm = fmaxf(tm, t[i]);
        tm = fmaxf(tm, __shfl_xor(tm, 16, 32));

        const float newm = fmaxf(mrun, tm);
        const float sc   = __builtin_amdgcn_exp2f(mrun - newm);  // raw v_exp_f32
        mrun = newm;

        // ---- p = 2^(s - m) (Q pre-scaled by log2e), sum with ONE exchange ---
        float rs = 0.0f;
#pragma unroll
        for (int nt = 0; nt < 4; ++nt)
#pragma unroll
            for (int i = 0; i < 8; ++i) {
                const float p = __builtin_amdgcn_exp2f(s[nt][i] - newm);
                s[nt][i] = p;
                rs += p;
            }
        rs += __shfl_xor(rs, 16, 32);
        lrun = lrun * sc + rs;
#pragma unroll
        for (int dt = 0; dt < 4; ++dt)
#pragma unroll
            for (int i = 0; i < 8; ++i) oacc[dt][i] *= sc;

        // ---- P^T -> B-fragment via wave-private LDS (4 x b128 each way) -----
#pragma unroll
        for (int nt = 0; nt < 4; ++nt) {           // lane's 8 p's are contiguous in n
            U8 u;
#pragma unroll
            for (int j = 0; j < 4; ++j)
                u.p[j] = __builtin_amdgcn_cvt_pkrtz(s[nt][2 * j], s[nt][2 * j + 1]);
            *(v8h*)&Pw[lh * PSTR + nt * 16 + hi * 8] = u.v;
        }
        U16 pb[2];
#pragma unroll
        for (int kt = 0; kt < 2; ++kt) {           // B-frag: contiguous 32B per lane
            const _Float16* pp = &Pw[lh * PSTR + kt * 32 + hi * 16];
            pb[kt].h[0] = *(const v8h*)(pp);
            pb[kt].h[1] = *(const v8h*)(pp + 8);
        }

        // ---- O^T += V^T x P^T ----------------------------------------------
#pragma unroll
        for (int dt = 0; dt < 4; ++dt) {
            const _Float16* vr = vbase + (size_t)(dt * 16 + lh) * LK_ + kv;
#pragma unroll
            for (int kt = 0; kt < 2; ++kt) {
                U16 va;
                const _Float16* c0 = vr + kt * 32 + hi * 16;
                va.h[0] = *(const v8h*)(c0);
                va.h[1] = *(const v8h*)(c0 + 8);
                oacc[dt] = __builtin_amdgcn_wmma_f32_16x16x32_f16(
                        false, va.v, false, pb[kt].v, (short)0, oacc[dt], false, false);
            }
        }
    }

    // ---- epilogue: per-lane row, contiguous 32B stores ----------------------
    const float inv = 1.0f / lrun;
    float* orow = O + ((size_t)batch * LQ_ + qbase + wave * 16 + lh) * D_;
#pragma unroll
    for (int dt = 0; dt < 4; ++dt) {
        f32x4 o0, o1;
#pragma unroll
        for (int i = 0; i < 4; ++i) {
            o0[i] = oacc[dt][i]     * inv;
            o1[i] = oacc[dt][i + 4] * inv;
        }
        *(f32x4*)(orow + dt * 16 + hi * 8)     = o0;
        *(f32x4*)(orow + dt * 16 + hi * 8 + 4) = o1;
    }
}

extern "C" void kernel_launch(void* const* d_in, const int* in_sizes, int n_in,
                              void* d_out, int out_size, void* d_ws, size_t ws_size,
                              hipStream_t stream) {
    const float* Q  = (const float*)d_in[0];
    const float* K  = (const float*)d_in[1];
    const float* V  = (const float*)d_in[2];
    const int*   VL = (const int*)d_in[3];

    // workspace layout: Qh | Kh | Vt  (3 x 4 MiB of f16 = 12.6 MB)
    const size_t nElem = (size_t)B_ * LQ_ * D_;
    _Float16* Qh = (_Float16*)d_ws;
    _Float16* Kh = Qh + nElem;
    _Float16* Vt = Kh + nElem;

    const int cvtBlocks = (int)(nElem / (256 * 8));            // 1024
    cvt_f32_f16<<<cvtBlocks, 256, 0, stream>>>(Q, Qh, QSCALE); // Q pre-scaled
    cvt_f32_f16<<<cvtBlocks, 256, 0, stream>>>(K, Kh, 1.0f);
    transpose_v<<<B_ * (LK_ / 64), 256, 0, stream>>>(V, Vt);   // 512 blocks

    fa_fwd<<<B_ * (LQ_ / 128), 256, 0, stream>>>(Qh, Kh, Vt, VL, (float*)d_out);
}
